// EdgePredictionModel_74612171866597
// MI455X (gfx1250) — matrix operand order repaired and checked by reading, not compiled, not run on hardware
//
#include <hip/hip_runtime.h>

typedef __attribute__((ext_vector_type(16))) _Float16 v16h;
typedef __attribute__((ext_vector_type(8)))  _Float16 v8h;
typedef __attribute__((ext_vector_type(8)))  float    v8f;

#define HH 128  // hidden dim (fixed by the model)

// ---------------- f32 -> f16 convert ----------------
__global__ void sage_cvt_f16(const float* __restrict__ src, _Float16* __restrict__ dst, int n) {
    int i = blockIdx.x * blockDim.x + threadIdx.x;
    if (i < n) dst[i] = (_Float16)src[i];
}

// gather x = emb[node_id] directly into f16
__global__ void sage_gather_f16(const float* __restrict__ emb, const int* __restrict__ node_id,
                                _Float16* __restrict__ dst, int total) {
    int i = blockIdx.x * blockDim.x + threadIdx.x;
    if (i >= total) return;
    int row = i >> 7, col = i & 127;
    dst[i] = (_Float16)emb[(size_t)node_id[row] * HH + col];
}

// ---------------- scatter-add via packed f16 atomics ----------------
// one wave of 32 lanes per edge; each lane handles 4 halves = 2 pk atomics
// (GLOBAL_ATOMIC_PK_ADD_F16: one 32-bit atomic updates two f16 values -> halves
//  the atomic op count vs f32 scatter)
__global__ void sage_scatter_pk(const _Float16* __restrict__ x, const int* __restrict__ src,
                                const int* __restrict__ dst, _Float16* __restrict__ agg,
                                float* __restrict__ cnt, int E, int addCnt) {
    int t = blockIdx.x * blockDim.x + threadIdx.x;
    int e = t >> 5, lane = t & 31;
    if (e >= E) return;
    const uint2 v = *(const uint2*)(x + (size_t)src[e] * HH + lane * 4);
    unsigned long long a0 =
        (unsigned long long)(uintptr_t)(agg + (size_t)dst[e] * HH + lane * 4);
    asm volatile("global_atomic_pk_add_f16 %0, %1, off\n\t"
                 "global_atomic_pk_add_f16 %0, %2, off offset:4"
                 :
                 : "v"(a0), "v"(v.x), "v"(v.y)
                 : "memory");
    if (addCnt && lane == 0) unsafeAtomicAdd(cnt + dst[e], 1.0f);
}

// ---------------- mean finalize: agg / max(cnt,1) -> f16 ----------------
__global__ void sage_mean(const _Float16* __restrict__ agg, const float* __restrict__ cnt,
                          _Float16* __restrict__ mean_h, int total) {
    int i = blockIdx.x * blockDim.x + threadIdx.x;
    if (i >= total) return;
    float c = cnt[i >> 7];
    float d = c > 1.0f ? c : 1.0f;
    mean_h[i] = (_Float16)((float)agg[i] / d);
}

// ---------------- fused SAGE GEMM via WMMA ----------------
// out = mean @ Wl^T + self @ Wr^T + bias   (+optional ReLU)
// One wave per 16x16 output tile. 8 x v_wmma_f32_16x16x32_f16 per tile (K=128 twice).
__global__ __launch_bounds__(256) void sage_gemm_wmma(
    const _Float16* __restrict__ Amean, const _Float16* __restrict__ Aself,
    const _Float16* __restrict__ Wl,    const _Float16* __restrict__ Wr,
    const float* __restrict__ bias,
    float* __restrict__ outF, _Float16* __restrict__ outH,
    int relu, int N) {
    const int lane = threadIdx.x & 31;
    const int tn   = threadIdx.x >> 5;         // 8 column tiles per block
    const int tm   = blockIdx.x;
    const int m0 = tm * 16, n0 = tn * 16;
    const int lm = lane & 15;                  // M (A) or N (B,C) index
    const int hi = lane >> 4;                  // half-wave selector

    int arow = m0 + lm; if (arow > N - 1) arow = N - 1;   // clamp for ragged tail
    const _Float16* am = Amean + (size_t)arow * HH;
    const _Float16* as = Aself + (size_t)arow * HH;
    const _Float16* bl = Wl + (size_t)(n0 + lm) * HH;     // B[k][n] = W[n][k]
    const _Float16* br = Wr + (size_t)(n0 + lm) * HH;
    const int aoff = hi * 8;    // A frag: halves 0..7 -> K=aoff.., halves 8..15 -> K=aoff+16..
    const int boff = hi * 8;    // B frag: halves 0..7 -> K=boff.., halves 8..15 -> K=boff+16..

    v8f acc = {};
#pragma unroll
    for (int kt = 0; kt < 4; ++kt) {
        const int k0 = kt * 32;
        v16h a, b;
        *((v8h*)&a)     = *(const v8h*)(am + k0 + aoff);
        *((v8h*)&a + 1) = *(const v8h*)(am + k0 + aoff + 16);
        *((v8h*)&b)     = *(const v8h*)(bl + k0 + boff);
        *((v8h*)&b + 1) = *(const v8h*)(bl + k0 + boff + 16);
        acc = __builtin_amdgcn_wmma_f32_16x16x32_f16(false, a, false, b, (short)0, acc, false, false);
    }
#pragma unroll
    for (int kt = 0; kt < 4; ++kt) {
        const int k0 = kt * 32;
        v16h a, b;
        *((v8h*)&a)     = *(const v8h*)(as + k0 + aoff);
        *((v8h*)&a + 1) = *(const v8h*)(as + k0 + aoff + 16);
        *((v8h*)&b)     = *(const v8h*)(br + k0 + boff);
        *((v8h*)&b + 1) = *(const v8h*)(br + k0 + boff + 16);
        acc = __builtin_amdgcn_wmma_f32_16x16x32_f16(false, a, false, b, (short)0, acc, false, false);
    }

    const float bn = bias[n0 + lm];
#pragma unroll
    for (int r = 0; r < 8; ++r) {
        int m = m0 + hi * 8 + r;
        if (m >= N) continue;
        float v = acc[r] + bn;
        if (relu) v = v > 0.0f ? v : 0.0f;
        size_t o = (size_t)m * HH + n0 + lm;
        if (outF) outF[o] = v;
        if (outH) outH[o] = (_Float16)v;
    }
}

// ---------------- edge scorer: dot(h[u], h[v]) ----------------
__global__ void sage_score(const float* __restrict__ h, const int* __restrict__ u,
                           const int* __restrict__ v, float* __restrict__ out, int L) {
    int t = blockIdx.x * blockDim.x + threadIdx.x;
    int e = t >> 5, lane = t & 31;
    if (e >= L) return;
    const float4 a = *(const float4*)(h + (size_t)u[e] * HH + lane * 4);
    const float4 b = *(const float4*)(h + (size_t)v[e] * HH + lane * 4);
    float s = a.x * b.x + a.y * b.y + a.z * b.z + a.w * b.w;
#pragma unroll
    for (int off = 16; off > 0; off >>= 1) s += __shfl_down(s, off, 32);
    if (lane == 0) out[e] = s;
}

extern "C" void kernel_launch(void* const* d_in, const int* in_sizes, int n_in,
                              void* d_out, int out_size, void* d_ws, size_t ws_size,
                              hipStream_t stream) {
    const int*   node_id = (const int*)d_in[0];
    const int*   ei      = (const int*)d_in[1];   // [2, E]
    const int*   eli     = (const int*)d_in[2];   // [2, L]
    const float* emb     = (const float*)d_in[3];
    const float* Wl1     = (const float*)d_in[4];
    const float* Wr1     = (const float*)d_in[5];
    const float* b1      = (const float*)d_in[6];
    const float* Wl2     = (const float*)d_in[7];
    const float* Wr2     = (const float*)d_in[8];
    const float* b2      = (const float*)d_in[9];
    float* out = (float*)d_out;

    const int N = in_sizes[0];
    const int E = in_sizes[1] / 2;
    const int L = in_sizes[2] / 2;
    const size_t NH = (size_t)N * HH;

    // workspace carve-out (256B aligned slices)
    char* p = (char*)d_ws;
    auto carve = [&](size_t bytes) -> char* {
        char* r = p; p += (bytes + 255) & ~(size_t)255; return r;
    };
    _Float16* agg    = (_Float16*)carve(NH * 2);
    float*    cnt    = (float*)   carve((size_t)N * 4);
    _Float16* mean_h = (_Float16*)carve(NH * 2);
    _Float16* x_h    = (_Float16*)carve(NH * 2);
    _Float16* h1_h   = (_Float16*)carve(NH * 2);
    float*    h2     = (float*)   carve(NH * 4);
    _Float16* wl1h   = (_Float16*)carve((size_t)HH * HH * 2);
    _Float16* wr1h   = (_Float16*)carve((size_t)HH * HH * 2);
    _Float16* wl2h   = (_Float16*)carve((size_t)HH * HH * 2);
    _Float16* wr2h   = (_Float16*)carve((size_t)HH * HH * 2);

    const int TPB = 256;
    const int wtot = HH * HH;
    const int gw = (wtot + TPB - 1) / TPB;
    sage_cvt_f16<<<gw, TPB, 0, stream>>>(Wl1, wl1h, wtot);
    sage_cvt_f16<<<gw, TPB, 0, stream>>>(Wr1, wr1h, wtot);
    sage_cvt_f16<<<gw, TPB, 0, stream>>>(Wl2, wl2h, wtot);
    sage_cvt_f16<<<gw, TPB, 0, stream>>>(Wr2, wr2h, wtot);

    const int gnh = (int)((NH + TPB - 1) / TPB);
    sage_gather_f16<<<gnh, TPB, 0, stream>>>(emb, node_id, x_h, (int)NH);

    hipMemsetAsync(agg, 0, NH * 2, stream);
    hipMemsetAsync(cnt, 0, (size_t)N * 4, stream);

    const int gedge = (int)(((size_t)E * 32 + TPB - 1) / TPB);
    sage_scatter_pk<<<gedge, TPB, 0, stream>>>(x_h, ei, ei + E, agg, cnt, E, /*addCnt=*/1);
    sage_mean<<<gnh, TPB, 0, stream>>>(agg, cnt, mean_h, (int)NH);

    const int gtm = (N + 15) / 16;   // 8 waves/block cover all 8 column tiles
    sage_gemm_wmma<<<gtm, 256, 0, stream>>>(mean_h, x_h, wl1h, wr1h, b1,
                                            nullptr, h1_h, /*relu=*/1, N);

    hipMemsetAsync(agg, 0, NH * 2, stream);   // counts are reused (same graph)
    sage_scatter_pk<<<gedge, TPB, 0, stream>>>(h1_h, ei, ei + E, agg, cnt, E, /*addCnt=*/0);
    sage_mean<<<gnh, TPB, 0, stream>>>(agg, cnt, mean_h, (int)NH);

    sage_gemm_wmma<<<gtm, 256, 0, stream>>>(mean_h, h1_h, wl2h, wr2h, b2,
                                            h2, nullptr, /*relu=*/0, N);

    const int gsc = (int)(((size_t)L * 32 + TPB - 1) / TPB);
    sage_score<<<gsc, TPB, 0, stream>>>(h2, eli, eli + L, out, L);
}
